// StackedAttentionModel_50689204027882
// MI455X (gfx1250) — compile-verified, matrix operand order checked by reading, not run on hardware
//
#include <hip/hip_runtime.h>
#include <hip/hip_bf16.h>

// ---------- model dims ----------
#define BB   128
#define LL   20
#define EE   256
#define CC   2048
#define KK   400
#define NCC  2000
#define HH   1024
#define GG   3072          // 3*H
#define PP   196           // 14*14
#define BL   (BB*LL)       // 2560
#define MIMG (BB*PP)       // 25088

typedef __attribute__((ext_vector_type(16))) _Float16 v16h;
typedef __attribute__((ext_vector_type(8)))  _Float16 v8h;
typedef __attribute__((ext_vector_type(8)))  float    v8f;

// ---------- small helpers ----------
__device__ inline float wave_sum(float v) {
  #pragma unroll
  for (int off = 16; off > 0; off >>= 1) v += __shfl_xor(v, off, 32);
  return v;
}
__device__ inline float sigmoidf_(float x) { return 1.0f / (1.0f + __expf(-x)); }

// async copy of 16 bytes global -> LDS, tracked by ASYNCcnt (CDNA5 path)
__device__ inline void async_b128_to_lds(const void* gsrc, const _Float16* lds_dst) {
  unsigned ldsa = (unsigned)(uintptr_t)lds_dst;   // low 32 bits = LDS byte offset
  asm volatile("global_load_async_to_lds_b128 %0, %1, off"
               :: "v"(ldsa), "v"(gsrc) : "memory");
}
__device__ inline void wait_asynccnt0() {
  asm volatile("s_wait_asynccnt 0x0" ::: "memory");
}

// ---------- f32 -> f16 convert ----------
__global__ void k_cvt(const float* __restrict__ s, _Float16* __restrict__ d, int n) {
  int i = blockIdx.x * blockDim.x + threadIdx.x;
  if (i < n) d[i] = (_Float16)s[i];
}

// ---------- zero fill (f32 + f16 pair) ----------
__global__ void k_zero(float* __restrict__ a, _Float16* __restrict__ b, int n) {
  int i = blockIdx.x * blockDim.x + threadIdx.x;
  if (i < n) { a[i] = 0.0f; b[i] = (_Float16)0.0f; }
}

// ---------- embedding gather: x16[b*L+l][e] = emb[question[b*L+l]][e] ----------
__global__ void k_embed(const int* __restrict__ q, const float* __restrict__ emb,
                        _Float16* __restrict__ x16) {
  int row = blockIdx.x;            // 0..BL-1
  int e   = threadIdx.x;           // 0..255
  int tok = q[row];
  x16[(size_t)row * EE + e] = (_Float16)emb[(size_t)tok * EE + e];
}

// ---------- image transpose: [B,C,P] f32 -> [B*P, C] f16 (LDS tiled) ----------
__global__ void k_transpose_img(const float* __restrict__ img, _Float16* __restrict__ out) {
  __shared__ float tile[32][33];
  int b  = blockIdx.z;
  int c0 = blockIdx.x * 32;
  int p0 = blockIdx.y * 32;
  int tx = threadIdx.x;            // 0..31
  int ty = threadIdx.y;            // 0..7
  #pragma unroll
  for (int i = 0; i < 32; i += 8) {
    int c = c0 + ty + i, p = p0 + tx;
    tile[ty + i][tx] = (p < PP) ? img[((size_t)b * CC + c) * PP + p] : 0.0f;
  }
  __syncthreads();
  #pragma unroll
  for (int i = 0; i < 32; i += 8) {
    int p = p0 + ty + i, c = c0 + tx;
    if (p < PP) out[((size_t)b * PP + p) * CC + c] = (_Float16)tile[tx][ty + i];
  }
}

// ---------- WMMA GEMM with async-LDS staged B panel -------------------------
// C[M,N] = A[M,K](f16) * B[N,K]^T(f16) + bias[N]
// Block = 256 threads = 8 waves. Block tile: 128 rows x (16*NT) cols; wave w
// owns rows [m0+16w, m0+16w+16). All 8 waves share a B panel (16*NT cols x 32 K
// halves) staged in LDS via global_load_async_to_lds_b128, double-buffered so
// the copy of chunk k+1 overlaps the WMMAs of chunk k (ASYNCcnt handoff).
// Per iteration: A global load issued before the barrier (stays in flight),
// then all NT B fragments are pulled from LDS in one ds clause, then the NT
// WMMAs issue back-to-back (XDL-friendly).
// Requires: M % 128 == 0, N % (16*NT) == 0, K % 32 == 0.
// CDNA5 16-bit fragment layout: lanes 0-15 hold K {k0..k0+7, k0+16..k0+23},
// lanes 16-31 hold K {k0+8..k0+15, k0+24..k0+31} for their row (A) / col (B).
template<int NT>
__global__ __launch_bounds__(256)
void k_gemm_f16(const _Float16* __restrict__ A, const _Float16* __restrict__ Bw,
                float* __restrict__ C, const float* __restrict__ bias,
                int M, int N, int K) {
  __shared__ _Float16 bsh[2][NT * 16 * 32];   // [buf][col][32 K-halves]
  int tid  = threadIdx.x;
  int lane = tid & 31;
  int wv   = tid >> 5;                        // 0..7
  int nblk = N / (16 * NT);
  int bm   = blockIdx.x / nblk;
  int bn   = blockIdx.x % nblk;
  int m0   = bm * 128 + wv * 16;
  int n0   = bn * (16 * NT);

  int lrow  = lane & 15;
  int khalf = (lane >> 4) << 3;               // 0 or 8

  v8f acc[NT];
  #pragma unroll
  for (int t = 0; t < NT; ++t)
    #pragma unroll
    for (int i = 0; i < 8; ++i) acc[t][i] = 0.0f;

  const _Float16* arow = A + (size_t)(m0 + lrow) * K + khalf;

  // prologue: stage B chunk 0 into buffer 0
  #pragma unroll 1
  for (int idx = tid; idx < NT * 64; idx += 256) {         // 16B granules
    int col = idx >> 2, part = idx & 3;
    async_b128_to_lds(Bw + (size_t)(n0 + col) * K + part * 8,
                      &bsh[0][col * 32 + part * 8]);
  }

  int nk = K >> 5;
  for (int ki = 0; ki < nk; ++ki) {
    int cur = ki & 1;
    int k0  = ki << 5;

    // A fragment global load: issue early, stays in flight across the
    // asynccnt wait + barrier + async staging below.
    v8h alo = *(const v8h*)(arow + k0);
    v8h ahi = *(const v8h*)(arow + k0 + 16);
    if (k0 + 64 <= K) __builtin_prefetch(arow + k0 + 64);  // global_prefetch_b8

    wait_asynccnt0();          // my async writes for buf 'cur' done
    __syncthreads();           // everyone's writes visible; prev buf fully consumed
    if (ki + 1 < nk) {         // stage next chunk; overlaps with compute below
      int k0n = (ki + 1) << 5;
      #pragma unroll 1
      for (int idx = tid; idx < NT * 64; idx += 256) {
        int col = idx >> 2, part = idx & 3;
        async_b128_to_lds(Bw + (size_t)(n0 + col) * K + k0n + part * 8,
                          &bsh[1 - cur][col * 32 + part * 8]);
      }
    }

    // pull ALL B fragments first (one ds clause, one wait) ...
    v16h bf[NT];
    #pragma unroll
    for (int t = 0; t < NT; ++t) {
      int lcol = t * 16 + lrow;
      v8h blo = *(const v8h*)&bsh[cur][lcol * 32 + khalf];         // ds_load_b128
      v8h bhi = *(const v8h*)&bsh[cur][lcol * 32 + khalf + 16];
      bf[t] = __builtin_shufflevector(blo, bhi, 0,1,2,3,4,5,6,7,8,9,10,11,12,13,14,15);
    }
    v16h a = __builtin_shufflevector(alo, ahi, 0,1,2,3,4,5,6,7,8,9,10,11,12,13,14,15);

    // ... then issue the WMMAs back-to-back (independent accumulators)
    #pragma unroll
    for (int t = 0; t < NT; ++t) {
      acc[t] = __builtin_amdgcn_wmma_f32_16x16x32_f16(
          /*neg_a=*/false, a, /*neg_b=*/false, bf[t],
          /*c_mod=*/(short)0, acc[t], /*reuse_a=*/false, /*reuse_b=*/false);
    }
  }

  // f32 C/D layout: VGPR i -> row (i + 8*(lane>>4)), col lane&15
  int r0 = (lane >> 4) << 3;
  int cc = lane & 15;
  #pragma unroll
  for (int t = 0; t < NT; ++t) {
    int coln = n0 + t * 16 + cc;
    float bv = bias ? bias[coln] : 0.0f;
    #pragma unroll
    for (int i = 0; i < 8; ++i)
      C[(size_t)(m0 + r0 + i) * N + coln] = acc[t][i] + bv;
  }
}

// ---------- GRU gate fusion: h = (1-z)*n + z*h (in place), also emit f16 h ----------
__global__ void k_gru_gate(const float* __restrict__ xp, const float* __restrict__ gh,
                           float* __restrict__ h, _Float16* __restrict__ h16, int l) {
  int i = blockIdx.x * blockDim.x + threadIdx.x;      // 0 .. B*H-1
  if (i >= BB * HH) return;
  int b = i >> 10, j = i & (HH - 1);
  const float* xr  = xp + (size_t)(b * LL + l) * GG;  // row includes b_ih already
  const float* ghb = gh + (size_t)b * GG;             // includes b_hh already
  float r = sigmoidf_(xr[j]          + ghb[j]);
  float z = sigmoidf_(xr[HH + j]     + ghb[HH + j]);
  float n = tanhf    (xr[2 * HH + j] + r * ghb[2 * HH + j]);
  float hv = (1.0f - z) * n + z * h[i];
  h[i] = hv;
  h16[i] = (_Float16)hv;
}

// ---------- concat q = [h_f, h_b] ----------
__global__ void k_concat(const float* __restrict__ hf, const float* __restrict__ hb,
                         float* __restrict__ q, _Float16* __restrict__ q16) {
  int i = blockIdx.x * blockDim.x + threadIdx.x;      // 0 .. B*C-1
  if (i >= BB * CC) return;
  int b = i >> 11, j = i & (CC - 1);
  float v = (j < HH) ? hf[b * HH + j] : hb[b * HH + (j - HH)];
  q[i] = v; q16[i] = (_Float16)v;
}

// ---------- attention score: e[b,p] = sum_j tanh(di[bp,j]+dq[b,j])*We[j] + be ----------
__global__ void k_att_score(const float* __restrict__ di, const float* __restrict__ dq,
                            const float* __restrict__ We, const float* __restrict__ be,
                            float* __restrict__ e) {
  int wave = (int)((blockIdx.x * blockDim.x + threadIdx.x) >> 5);   // 0..MIMG-1
  int lane = threadIdx.x & 31;
  if (wave >= MIMG) return;
  int b = wave / PP;
  const float* dr  = di + (size_t)wave * KK;
  const float* dqb = dq + (size_t)b * KK;
  float s = 0.0f;
  for (int j = lane; j < KK; j += 32) s += tanhf(dr[j] + dqb[j]) * We[j];
  s = wave_sum(s);
  if (lane == 0) e[wave] = s + be[0];
}

// ---------- softmax over P per batch ----------
__global__ void k_softmax(const float* __restrict__ e, float* __restrict__ alpha) {
  __shared__ float red[256];
  int b = blockIdx.x, t = threadIdx.x;
  float v = (t < PP) ? e[b * PP + t] : -1e30f;
  red[t] = v; __syncthreads();
  for (int s = 128; s > 0; s >>= 1) { if (t < s) red[t] = fmaxf(red[t], red[t + s]); __syncthreads(); }
  float mx = red[0]; __syncthreads();
  float ex = (t < PP) ? __expf(v - mx) : 0.0f;
  red[t] = ex; __syncthreads();
  for (int s = 128; s > 0; s >>= 1) { if (t < s) red[t] += red[t + s]; __syncthreads(); }
  float inv = 1.0f / red[0];
  if (t < PP) alpha[b * PP + t] = ex * inv;
}

// ---------- attention pool: u[b,c] = sum_p image[b,c,p]*alpha[b,p] + resid[b,c] ----------
__global__ void k_att_pool(const float* __restrict__ img, const float* __restrict__ alpha,
                           const float* __restrict__ resid,
                           float* __restrict__ u, _Float16* __restrict__ u16) {
  int wave = (int)((blockIdx.x * blockDim.x + threadIdx.x) >> 5);   // b*C + c
  int lane = threadIdx.x & 31;
  if (wave >= BB * CC) return;
  int b = wave >> 11;
  const float* ic = img + (size_t)wave * PP;
  const float* al = alpha + (size_t)b * PP;
  float s = 0.0f;
  for (int p = lane; p < PP; p += 32) s += ic[p] * al[p];
  s = wave_sum(s);
  if (lane == 0) { float v = s + resid[wave]; u[wave] = v; u16[wave] = (_Float16)v; }
}

// ---------- log_softmax over NC per batch ----------
__global__ void k_logsoftmax(const float* __restrict__ logits, float* __restrict__ out) {
  __shared__ float red[256];
  int b = blockIdx.x, t = threadIdx.x;
  float mx = -1e30f;
  for (int j = t; j < NCC; j += 256) mx = fmaxf(mx, logits[b * NCC + j]);
  red[t] = mx; __syncthreads();
  for (int s = 128; s > 0; s >>= 1) { if (t < s) red[t] = fmaxf(red[t], red[t + s]); __syncthreads(); }
  mx = red[0]; __syncthreads();
  float sm = 0.0f;
  for (int j = t; j < NCC; j += 256) sm += __expf(logits[b * NCC + j] - mx);
  red[t] = sm; __syncthreads();
  for (int s = 128; s > 0; s >>= 1) { if (t < s) red[t] += red[t + s]; __syncthreads(); }
  float lse = mx + __logf(red[0]);
  for (int j = t; j < NCC; j += 256) out[b * NCC + j] = logits[b * NCC + j] - lse;
}

// =======================================================================
extern "C" void kernel_launch(void* const* d_in, const int* in_sizes, int n_in,
                              void* d_out, int out_size, void* d_ws, size_t ws_size,
                              hipStream_t stream) {
  (void)in_sizes; (void)n_in; (void)out_size; (void)ws_size;
  const int*   question = (const int*)  d_in[0];
  const float* image    = (const float*)d_in[1];
  const float* emb      = (const float*)d_in[2];
  const float* W_ih_f   = (const float*)d_in[3];
  const float* W_hh_f   = (const float*)d_in[4];
  const float* b_ih_f   = (const float*)d_in[5];
  const float* b_hh_f   = (const float*)d_in[6];
  const float* W_ih_b   = (const float*)d_in[7];
  const float* W_hh_b   = (const float*)d_in[8];
  const float* b_ih_b   = (const float*)d_in[9];
  const float* b_hh_b   = (const float*)d_in[10];
  const float* Wi1 = (const float*)d_in[11]; const float* bi1 = (const float*)d_in[12];
  const float* Wq1 = (const float*)d_in[13]; const float* We1 = (const float*)d_in[14];
  const float* be1 = (const float*)d_in[15];
  const float* Wi2 = (const float*)d_in[16]; const float* bi2 = (const float*)d_in[17];
  const float* Wq2 = (const float*)d_in[18]; const float* We2 = (const float*)d_in[19];
  const float* be2 = (const float*)d_in[20];
  const float* Wfc = (const float*)d_in[21]; const float* bfc = (const float*)d_in[22];
  float* out = (float*)d_out;

  // ---- workspace carve-up ----
  char* w = (char*)d_ws; size_t off = 0;
  auto alloc = [&](size_t bytes) -> void* {
    void* p = w + off; off += (bytes + 255) & ~(size_t)255; return p;
  };
  _Float16* wihf16 = (_Float16*)alloc((size_t)GG * EE * 2);
  _Float16* wihb16 = (_Float16*)alloc((size_t)GG * EE * 2);
  _Float16* whhf16 = (_Float16*)alloc((size_t)GG * HH * 2);
  _Float16* whhb16 = (_Float16*)alloc((size_t)GG * HH * 2);
  _Float16* wi116  = (_Float16*)alloc((size_t)KK * CC * 2);
  _Float16* wq116  = (_Float16*)alloc((size_t)KK * CC * 2);
  _Float16* wi216  = (_Float16*)alloc((size_t)KK * CC * 2);
  _Float16* wq216  = (_Float16*)alloc((size_t)KK * CC * 2);
  _Float16* wfc16  = (_Float16*)alloc((size_t)NCC * CC * 2);
  _Float16* x16    = (_Float16*)alloc((size_t)BL * EE * 2);
  _Float16* img16  = (_Float16*)alloc((size_t)MIMG * CC * 2);
  float*    xp_f   = (float*)   alloc((size_t)BL * GG * 4);
  float*    xp_b   = (float*)   alloc((size_t)BL * GG * 4);
  float*    gh     = (float*)   alloc((size_t)BB * GG * 4);
  float*    h32f   = (float*)   alloc((size_t)BB * HH * 4);
  float*    h32b   = (float*)   alloc((size_t)BB * HH * 4);
  _Float16* h16f   = (_Float16*)alloc((size_t)BB * HH * 2);
  _Float16* h16b   = (_Float16*)alloc((size_t)BB * HH * 2);
  float*    q32    = (float*)   alloc((size_t)BB * CC * 4);
  _Float16* q16    = (_Float16*)alloc((size_t)BB * CC * 2);
  float*    dq     = (float*)   alloc((size_t)BB * KK * 4);
  float*    di     = (float*)   alloc((size_t)MIMG * KK * 4);
  float*    esc    = (float*)   alloc((size_t)BB * PP * 4);
  float*    alpha  = (float*)   alloc((size_t)BB * PP * 4);
  float*    u1     = (float*)   alloc((size_t)BB * CC * 4);
  _Float16* u116   = (_Float16*)alloc((size_t)BB * CC * 2);
  float*    u2     = (float*)   alloc((size_t)BB * CC * 4);
  _Float16* u216   = (_Float16*)alloc((size_t)BB * CC * 2);
  float*    logits = (float*)   alloc((size_t)BB * NCC * 4);

  auto cvt = [&](const float* s, _Float16* d, int n) {
    k_cvt<<<(n + 255) / 256, 256, 0, stream>>>(s, d, n);
  };
  auto gemm4 = [&](const _Float16* A, const _Float16* Bm, float* Cm,
                   const float* bias, int M, int N, int K) {
    int blocks = (M / 128) * (N / 64);
    k_gemm_f16<4><<<blocks, 256, 0, stream>>>(A, Bm, Cm, bias, M, N, K);
  };
  auto gemm5 = [&](const _Float16* A, const _Float16* Bm, float* Cm,
                   const float* bias, int M, int N, int K) {
    int blocks = (M / 128) * (N / 80);
    k_gemm_f16<5><<<blocks, 256, 0, stream>>>(A, Bm, Cm, bias, M, N, K);
  };

  // 1) weights -> f16 (once per launch; deterministic)
  cvt(W_ih_f, wihf16, GG * EE);   cvt(W_ih_b, wihb16, GG * EE);
  cvt(W_hh_f, whhf16, GG * HH);   cvt(W_hh_b, whhb16, GG * HH);
  cvt(Wi1, wi116, KK * CC);       cvt(Wq1, wq116, KK * CC);
  cvt(Wi2, wi216, KK * CC);       cvt(Wq2, wq216, KK * CC);
  cvt(Wfc, wfc16, NCC * CC);

  // 2) embedding gather (f16 activations)
  k_embed<<<BL, EE, 0, stream>>>(question, emb, x16);

  // 3) image transpose [B,C,P] -> [B*P,C] f16 (LDS tiled)
  k_transpose_img<<<dim3(CC / 32, (PP + 31) / 32, BB), dim3(32, 8), 0, stream>>>(image, img16);

  // 4) input projections: xp = x @ W_ih^T + b_ih   [BL, 3H]
  gemm4(x16, wihf16, xp_f, b_ih_f, BL, GG, EE);
  gemm4(x16, wihb16, xp_b, b_ih_b, BL, GG, EE);

  // 5) bidirectional GRU (sequential scans; each step = WMMA hh-GEMM + fused gates)
  k_zero<<<(BB * HH + 255) / 256, 256, 0, stream>>>(h32f, h16f, BB * HH);
  for (int l = 0; l < LL; ++l) {
    gemm4(h16f, whhf16, gh, b_hh_f, BB, GG, HH);
    k_gru_gate<<<(BB * HH + 255) / 256, 256, 0, stream>>>(xp_f, gh, h32f, h16f, l);
  }
  k_zero<<<(BB * HH + 255) / 256, 256, 0, stream>>>(h32b, h16b, BB * HH);
  for (int l = LL - 1; l >= 0; --l) {
    gemm4(h16b, whhb16, gh, b_hh_b, BB, GG, HH);
    k_gru_gate<<<(BB * HH + 255) / 256, 256, 0, stream>>>(xp_b, gh, h32b, h16b, l);
  }
  k_concat<<<(BB * CC + 255) / 256, 256, 0, stream>>>(h32f, h32b, q32, q16);

  // 6) attention layer 1
  gemm5(q16,   wi116, dq, bi1,     BB,   KK, CC);   // dq = q @ Wi1^T + bi1
  gemm5(img16, wq116, di, nullptr, MIMG, KK, CC);   // di = img @ Wq1^T
  k_att_score<<<(MIMG * 32 + 255) / 256, 256, 0, stream>>>(di, dq, We1, be1, esc);
  k_softmax<<<BB, 256, 0, stream>>>(esc, alpha);
  k_att_pool<<<(BB * CC * 32 + 255) / 256, 256, 0, stream>>>(image, alpha, q32, u1, u116);

  // 7) attention layer 2
  gemm5(u116,  wi216, dq, bi2,     BB,   KK, CC);
  gemm5(img16, wq216, di, nullptr, MIMG, KK, CC);
  k_att_score<<<(MIMG * 32 + 255) / 256, 256, 0, stream>>>(di, dq, We2, be2, esc);
  k_softmax<<<BB, 256, 0, stream>>>(esc, alpha);
  k_att_pool<<<(BB * CC * 32 + 255) / 256, 256, 0, stream>>>(image, alpha, u1, u2, u216);

  // 8) classifier + log_softmax
  gemm5(u216, wfc16, logits, bfc, BB, NCC, CC);
  k_logsoftmax<<<BB, 256, 0, stream>>>(logits, out);
}
